// AttentionBlock_19215683682316
// MI455X (gfx1250) — matrix operand order repaired and checked by reading, not compiled, not run on hardware
//
#include <hip/hip_runtime.h>
#include <stdint.h>

#define BATCH 4
#define CCH   256
#define DQK   32
#define NSP   4096
#define TJ    32

typedef __attribute__((ext_vector_type(16))) __bf16        v16bf;
typedef __attribute__((ext_vector_type(8)))  float         v8f;
typedef __attribute__((ext_vector_type(4)))  unsigned int  u32x4;
typedef __attribute__((ext_vector_type(8)))  int           i32x8;
typedef __attribute__((ext_vector_type(4)))  int           i32x4;

union BF16Frag { v16bf v; u32x4 q[2]; unsigned short s[16]; };

__device__ __forceinline__ unsigned short f32_bf16(float x) {
  union { float f; uint32_t u; } a; a.f = x;
  uint32_t r = a.u + 0x7FFFu + ((a.u >> 16) & 1u);   // RNE
  return (unsigned short)(r >> 16);
}

// ---------------------------------------------------------------------------
// TDM: issue a 2-D bf16 tile load (tile_d0 x tile_d1, row stride d0_stride
// elements) from global into LDS at byte offset lds_addr. ISA 8.3-8.6 D#.
// 6-arg builtin flavor (clang-23 / therock-10.0 headers).
// ---------------------------------------------------------------------------
__device__ __forceinline__ void tdm_load_2d_bf16(
    unsigned lds_addr, const void* gptr,
    unsigned tile_d0, unsigned tile_d1,
    unsigned tensor_d0, unsigned tensor_d1,
    unsigned long long d0_stride)
{
  unsigned long long ga = (unsigned long long)(size_t)gptr;
  u32x4 g0;
  g0[0] = 1u;                                            // count=1, user desc
  g0[1] = lds_addr;                                      // LDS byte address
  g0[2] = (unsigned)(ga & 0xFFFFFFFFu);                  // global_addr[31:0]
  g0[3] = (unsigned)((ga >> 32) & 0x01FFFFFFu)           // global_addr[56:32]
        | (2u << 30);                                    // type=2 ("image")
  i32x8 g1;
  g1[0] = (int)(1u << 16);                               // mask=0, data_size=1 (2B)
  g1[1] = (int)((tensor_d0 & 0xFFFFu) << 16);            // barrier=0 | tdim0 lo16
  g1[2] = (int)(((tensor_d0 >> 16) & 0xFFFFu)
        | ((tensor_d1 & 0xFFFFu) << 16));                // tdim0 hi16 | tdim1 lo16
  g1[3] = (int)(((tensor_d1 >> 16) & 0xFFFFu)
        | ((tile_d0 & 0xFFFFu) << 16));                  // tdim1 hi16 | tile_dim0
  g1[4] = (int)(tile_d1 & 0xFFFFu);                      // tile_dim1 | tile_dim2=0
  g1[5] = (int)(d0_stride & 0xFFFFFFFFu);                // dim0_stride lo32
  g1[6] = (int)((d0_stride >> 32) & 0xFFFFu);            // dim0_stride hi16
  g1[7] = 0;
  i32x4 z4 = {0, 0, 0, 0};                               // groups 2/3: <=2D tensor
  i32x8 z8 = {0, 0, 0, 0, 0, 0, 0, 0};
  __builtin_amdgcn_tensor_load_to_lds(g0, g1, z4, z4, z8, 0);
}

// ---------------------------------------------------------------------------
// Kernel 1: q/k/v 1x1-conv projections -> bf16, packed for WMMA fragments.
//   qp, kp : [B][N][32]  (row-major over spatial pos, channels contiguous)
//   vp     : [B][C][N]   (channel-major, spatial contiguous)
// ---------------------------------------------------------------------------
__global__ __launch_bounds__(320) void qkv_proj_kernel(
    const float* __restrict__ x,
    const float* __restrict__ Wq, const float* __restrict__ bq,
    const float* __restrict__ Wk, const float* __restrict__ bk,
    const float* __restrict__ Wv, const float* __restrict__ bv,
    unsigned short* __restrict__ qp, unsigned short* __restrict__ kp,
    unsigned short* __restrict__ vp)
{
  __shared__ __align__(16) float xs[CCH][16];
  const int blk = blockIdx.x;
  const int b  = blk >> 8;          // 256 n-tiles per batch
  const int n0 = (blk & 255) * 16;
  const int t  = threadIdx.x;

  const float* xb = x + (size_t)b * CCH * NSP;
  for (int idx = t; idx < CCH * 16; idx += 320) {
    int c = idx >> 4, j = idx & 15;
    xs[c][j] = xb[(size_t)c * NSP + n0 + j];
  }
  __syncthreads();

  const float* wrow; float bias;
  if (t < 32)        { wrow = Wq + t * CCH;        bias = bq[t]; }
  else if (t < 64)   { wrow = Wk + (t - 32) * CCH; bias = bk[t - 32]; }
  else               { wrow = Wv + (t - 64) * CCH; bias = bv[t - 64]; }

  float acc[16];
  #pragma unroll
  for (int j = 0; j < 16; ++j) acc[j] = bias;
  for (int c = 0; c < CCH; ++c) {
    float w = wrow[c];
    #pragma unroll
    for (int j = 0; j < 16; ++j) acc[j] = fmaf(w, xs[c][j], acc[j]);
  }

  if (t < 64) {
    unsigned short* dst = (t < 32) ? qp : kp;
    int d = t & 31;
    #pragma unroll
    for (int j = 0; j < 16; ++j)
      dst[(size_t)(b * NSP + n0 + j) * DQK + d] = f32_bf16(acc[j]);
  } else {
    int c = t - 64;
    unsigned short* dst = vp + (size_t)(b * CCH + c) * NSP + n0;
    #pragma unroll
    for (int j = 0; j < 16; ++j) dst[j] = f32_bf16(acc[j]);
  }
}

// ---------------------------------------------------------------------------
// Kernel 2: flash-style attention, bf16 WMMA + TDM double-buffered staging.
// 4 waves/block; each wave owns a 16-row query tile. Per 32-key block:
//   S^T = WMMA(K_tile, Q)  (x2 subtiles) -> online softmax (per-lane stats)
//   O  += WMMA(V_tile, P^T)  for 16 channel tiles (K=32)
// Wave 0 drives the Tensor Data Mover; one barrier per iteration.
// ---------------------------------------------------------------------------
__global__ __launch_bounds__(128) void attn_kernel(
    const unsigned short* __restrict__ qp, const unsigned short* __restrict__ kp,
    const unsigned short* __restrict__ vp, const float* __restrict__ x,
    const float* __restrict__ gamma, float* __restrict__ out)
{
  __shared__ __align__(16) unsigned short klds[2][TJ][DQK];   // 2 x 2 KB
  __shared__ __align__(16) unsigned short vlds[2][CCH][TJ];   // 2 x 16 KB

  const int b     = blockIdx.x >> 6;       // 64 blocks per batch
  const int iblk  = blockIdx.x & 63;
  const int wave  = threadIdx.x >> 5;
  const int lane  = threadIdx.x & 31;
  const int iBase = iblk * 64 + wave * 16;
  const int col   = lane & 15;
  const int half  = lane >> 4;
  const bool lowh = (half == 0);

  const unsigned short* kpb = kp + (size_t)b * NSP * DQK;
  const unsigned short* vpb = vp + (size_t)b * CCH * NSP;
  const unsigned kbase = (unsigned)(size_t)&klds[0][0][0];   // LDS byte offsets
  const unsigned vbase = (unsigned)(size_t)&vlds[0][0][0];

  // Persistent Q B-fragment: lane=col i, d = half*8..+7 then 16+half*8..+7.
  BF16Frag qb;
  {
    const u32x4* src = (const u32x4*)(qp + (size_t)(b * NSP + iBase + col) * DQK + half * 8);
    qb.q[0] = src[0]; qb.q[1] = src[2];
  }

  v8f oacc[16];
  #pragma unroll
  for (int ct = 0; ct < 16; ++ct)
    #pragma unroll
    for (int p = 0; p < 8; ++p) oacc[ct][p] = 0.0f;

  float m = -1e30f, lsum = 0.0f;
  const v8f vzero = {0.f,0.f,0.f,0.f,0.f,0.f,0.f,0.f};

  // Prefetch tile 0 into buffer 0 (wave 0 drives the TDM).
  if (wave == 0) {
    tdm_load_2d_bf16(kbase, kpb, TJ * DQK, 1, NSP * DQK, 1, 0);        // 2KB linear
    tdm_load_2d_bf16(vbase, vpb, TJ, CCH, NSP, CCH, NSP);              // 256x32 tile
  }

  const int NT = NSP / TJ;
  for (int t = 0; t < NT; ++t) {
    const int cur = t & 1;
    if (wave == 0) __builtin_amdgcn_s_wait_tensorcnt(0);
    __syncthreads();   // tile t ready; all waves done computing tile t-1

    if (wave == 0 && (t + 1) < NT) {
      const int nxt = (t + 1) & 1;
      const int j1 = (t + 1) * TJ;
      tdm_load_2d_bf16(kbase + (unsigned)nxt * TJ * DQK * 2,
                       kpb + (size_t)j1 * DQK, TJ * DQK, 1, NSP * DQK, 1, 0);
      tdm_load_2d_bf16(vbase + (unsigned)nxt * CCH * TJ * 2,
                       vpb + j1, TJ, CCH, NSP, CCH, NSP);
    }

    // --- S^T tiles: rows j (two 16-row subtiles), cols i ---
    BF16Frag ka;
    ka.q[0] = *(const u32x4*)&klds[cur][col][half * 8];
    ka.q[1] = *(const u32x4*)&klds[cur][col][16 + half * 8];
    v8f s0 = __builtin_amdgcn_wmma_f32_16x16x32_bf16(false, ka.v, false, qb.v,
                                                     (short)0, vzero, false, false);
    ka.q[0] = *(const u32x4*)&klds[cur][16 + col][half * 8];
    ka.q[1] = *(const u32x4*)&klds[cur][16 + col][16 + half * 8];
    v8f s1 = __builtin_amdgcn_wmma_f32_16x16x32_bf16(false, ka.v, false, qb.v,
                                                     (short)0, vzero, false, false);

    // --- online softmax over keys (per-lane: fixed query column) ---
    float mx = -1e30f;
    #pragma unroll
    for (int p = 0; p < 8; ++p) { mx = fmaxf(mx, s0[p]); mx = fmaxf(mx, s1[p]); }
    mx = fmaxf(mx, __shfl_xor(mx, 16, 32));
    float mnew = fmaxf(m, mx);
    float corr = __expf(m - mnew);

    float t0[8], t1[8], rsum = 0.0f;
    #pragma unroll
    for (int p = 0; p < 8; ++p) {
      t0[p] = __expf(s0[p] - mnew);
      t1[p] = __expf(s1[p] - mnew);
      rsum += t0[p] + t1[p];
    }
    rsum += __shfl_xor(rsum, 16, 32);
    lsum = lsum * corr + rsum;
    m = mnew;

    #pragma unroll
    for (int ct = 0; ct < 16; ++ct)
      #pragma unroll
      for (int p = 0; p < 8; ++p) oacc[ct][p] *= corr;

    // --- rebuild P^T as WMMA B-fragment via cross-half shuffles ---
    float u0[8], u1[8];
    #pragma unroll
    for (int p = 0; p < 8; ++p) {
      u0[p] = __shfl_xor(t0[p], 16, 32);
      u1[p] = __shfl_xor(t1[p], 16, 32);
    }
    BF16Frag pb;
    #pragma unroll
    for (int e = 0; e < 8; ++e) {
      pb.s[e]     = f32_bf16(lowh ? t0[e] : u1[e]);   // K rows base+0..7
      pb.s[e + 8] = f32_bf16(lowh ? u0[e] : t1[e]);   // K rows base+8..15
    }

    // --- O += V_tile x P^T for all 16 channel tiles ---
    #pragma unroll
    for (int ct = 0; ct < 16; ++ct) {
      BF16Frag va;
      va.q[0] = *(const u32x4*)&vlds[cur][ct * 16 + col][half * 16];
      va.q[1] = *(const u32x4*)&vlds[cur][ct * 16 + col][half * 16 + 8];
      oacc[ct] = __builtin_amdgcn_wmma_f32_16x16x32_bf16(false, va.v, false, pb.v,
                                                         (short)0, oacc[ct], false, false);
    }
  }

  // --- epilogue: normalize, gamma, residual ---
  float inv = 1.0f / lsum;
  float g   = gamma[0];
  int   i   = iBase + col;
  #pragma unroll
  for (int ct = 0; ct < 16; ++ct) {
    #pragma unroll
    for (int p = 0; p < 8; ++p) {
      int c = ct * 16 + half * 8 + p;
      size_t off = (size_t)(b * CCH + c) * NSP + i;
      out[off] = fmaf(g, oacc[ct][p] * inv, x[off]);
    }
  }
}

// ---------------------------------------------------------------------------
extern "C" void kernel_launch(void* const* d_in, const int* in_sizes, int n_in,
                              void* d_out, int out_size, void* d_ws, size_t ws_size,
                              hipStream_t stream) {
  const float* x     = (const float*)d_in[0];
  const float* Wq    = (const float*)d_in[1];
  const float* bq    = (const float*)d_in[2];
  const float* Wk    = (const float*)d_in[3];
  const float* bk    = (const float*)d_in[4];
  const float* Wv    = (const float*)d_in[5];
  const float* bv    = (const float*)d_in[6];
  const float* gamma = (const float*)d_in[7];
  float* out = (float*)d_out;

  unsigned short* qp = (unsigned short*)d_ws;                    // 1 MB
  unsigned short* kp = qp + (size_t)BATCH * NSP * DQK;           // 1 MB
  unsigned short* vp = kp + (size_t)BATCH * NSP * DQK;           // 8 MB

  qkv_proj_kernel<<<BATCH * (NSP / 16), 320, 0, stream>>>(
      x, Wq, bq, Wk, bk, Wv, bv, qp, kp, vp);
  attn_kernel<<<BATCH * (NSP / 64), 128, 0, stream>>>(
      qp, kp, vp, x, gamma, out);
}
